// LowRankAttention_28561532519069
// MI455X (gfx1250) — compile-verified, matrix operand order checked by reading, not run on hardware
//
#include <hip/hip_runtime.h>
#include <hip/hip_bf16.h>

// ---------------------------------------------------------------------------
// Types for CDNA5 WMMA (wave32)
// ---------------------------------------------------------------------------
typedef __bf16 bf16_t;
typedef bf16_t bf16x16 __attribute__((ext_vector_type(16)));
typedef bf16_t bf16x8  __attribute__((ext_vector_type(8)));
typedef bf16_t bf16x4  __attribute__((ext_vector_type(4)));
typedef float  v8f     __attribute__((ext_vector_type(8)));
typedef float  v4f     __attribute__((ext_vector_type(4)));

// Native fp32 -> bf16 (RNE): lowers to hardware cvt on gfx1250
__device__ __forceinline__ bf16_t f2bf(float f) { return (bf16_t)f; }

__device__ __forceinline__ void store_bf4(bf16_t* p, float4 v) {
  v4f t = {v.x, v.y, v.z, v.w};
  *(bf16x4*)p = __builtin_convertvector(t, bf16x4);   // packed cvt + b64 store
}

__device__ __forceinline__ v8f zero_v8f() {
  v8f z;
#pragma unroll
  for (int i = 0; i < 8; ++i) z[i] = 0.0f;
  return z;
}

// A/B fragment for v_wmma_*_16x16x32: per-lane 16 bf16 elements, laid out as
// two contiguous 8-element chunks in LDS:  K = e + 8*((e>>3) + (lane>>4))
__device__ __forceinline__ bf16x16 load_frag(const bf16_t* p0, const bf16_t* p1) {
  bf16x8 c0 = *(const bf16x8*)p0;
  bf16x8 c1 = *(const bf16x8*)p1;
  bf16x16 f;
#pragma unroll
  for (int e = 0; e < 8; ++e) { f[e] = c0[e]; f[e + 8] = c1[e]; }
  return f;
}

// ---------------------------------------------------------------------------
// Generic tiled GEMM: C[M,N] = A[M,K] @ B[K,N] (+bias), fp32 in/out, bf16 WMMA.
// 256 threads = 8 waves; block tile 128x128; wave tile 32x64 (2x4 WMMA tiles).
// Requirements (met by all launches here): K % 32 == 0, lda/ldb/N % 4 == 0.
// Staging is branch-free (clamped address + value select) and float4-vector;
// next tile is register-double-buffered to overlap HBM/L2 latency with WMMA.
// ---------------------------------------------------------------------------
__global__ __launch_bounds__(256) void gemm_bf16_kernel(
    const float* __restrict__ A, const float* __restrict__ B,
    const float* __restrict__ bias, float* __restrict__ C,
    int M, int Nn, int K, int lda, int ldb, int ldc, int transB,
    int innerCount,
    long long aOuter, long long aInner,
    long long bOuter, long long bInner,
    long long cOuter, long long cInner)
{
  __shared__ __align__(16) bf16_t As[128 * 32];   // [m][k]
  __shared__ __align__(16) bf16_t Bs[128 * 32];   // [n][k]  (k contiguous)

  const int z = blockIdx.z;
  const long long aOff = (long long)(z / innerCount) * aOuter + (long long)(z % innerCount) * aInner;
  const long long bOff = (long long)(z / innerCount) * bOuter + (long long)(z % innerCount) * bInner;
  const long long cOff = (long long)(z / innerCount) * cOuter + (long long)(z % innerCount) * cInner;

  const int tid = threadIdx.x;
  const int lane = tid & 31;
  const int w = tid >> 5;
  const int wm = w & 3;        // wave row   (4 waves along M, 32 rows each)
  const int wn = w >> 2;       // wave col   (2 waves along N, 64 cols each)
  const int lo = lane & 15, hi = lane >> 4;
  const int blockM = blockIdx.y * 128;
  const int blockN = blockIdx.x * 128;

  v8f acc[2][4];
#pragma unroll
  for (int mt = 0; mt < 2; ++mt)
#pragma unroll
    for (int nt = 0; nt < 4; ++nt) acc[mt][nt] = zero_v8f();

  float4 aR[4], bR[4];  // register staging (double buffer vs. LDS)

  auto load_tiles = [&](int k0) {
    // A tile 128x32: 1024 float4, 4 per thread, contiguous along k
#pragma unroll
    for (int e = 0; e < 4; ++e) {
      int idx = tid + e * 256;
      int r = idx >> 3, c = (idx & 7) << 2;
      int gr = blockM + r;
      int grS = (gr < M) ? gr : 0;
      float4 v = *(const float4*)(A + aOff + (long long)grS * lda + (k0 + c));
      bool ok = gr < M;
      v.x = ok ? v.x : 0.0f; v.y = ok ? v.y : 0.0f;
      v.z = ok ? v.z : 0.0f; v.w = ok ? v.w : 0.0f;
      aR[e] = v;
    }
    if (!transB) {
      // memory [k][n]: float4 along n
#pragma unroll
      for (int e = 0; e < 4; ++e) {
        int idx = tid + e * 256;
        int kr = idx >> 5, n4 = (idx & 31) << 2;
        int gn = blockN + n4;
        int gnS = (gn < Nn) ? gn : 0;
        float4 v = *(const float4*)(B + bOff + (long long)(k0 + kr) * ldb + gnS);
        bool ok = gn < Nn;
        v.x = ok ? v.x : 0.0f; v.y = ok ? v.y : 0.0f;
        v.z = ok ? v.z : 0.0f; v.w = ok ? v.w : 0.0f;
        bR[e] = v;
      }
    } else {
      // memory [n][k]: float4 along k
#pragma unroll
      for (int e = 0; e < 4; ++e) {
        int idx = tid + e * 256;
        int nn = idx >> 3, k4 = (idx & 7) << 2;
        int gn = blockN + nn;
        int gnS = (gn < Nn) ? gn : 0;
        float4 v = *(const float4*)(B + bOff + (long long)gnS * ldb + (k0 + k4));
        bool ok = gn < Nn;
        v.x = ok ? v.x : 0.0f; v.y = ok ? v.y : 0.0f;
        v.z = ok ? v.z : 0.0f; v.w = ok ? v.w : 0.0f;
        bR[e] = v;
      }
    }
  };

  auto store_tiles = [&]() {
#pragma unroll
    for (int e = 0; e < 4; ++e) {
      int idx = tid + e * 256;
      int r = idx >> 3, c = (idx & 7) << 2;
      store_bf4(&As[r * 32 + c], aR[e]);
    }
    if (!transB) {
#pragma unroll
      for (int e = 0; e < 4; ++e) {
        int idx = tid + e * 256;
        int kr = idx >> 5, n4 = (idx & 31) << 2;
        Bs[(n4 + 0) * 32 + kr] = f2bf(bR[e].x);
        Bs[(n4 + 1) * 32 + kr] = f2bf(bR[e].y);
        Bs[(n4 + 2) * 32 + kr] = f2bf(bR[e].z);
        Bs[(n4 + 3) * 32 + kr] = f2bf(bR[e].w);
      }
    } else {
#pragma unroll
      for (int e = 0; e < 4; ++e) {
        int idx = tid + e * 256;
        int nn = idx >> 3, k4 = (idx & 7) << 2;
        store_bf4(&Bs[nn * 32 + k4], bR[e]);
      }
    }
  };

  load_tiles(0);
  for (int k0 = 0; k0 < K; k0 += 32) {
    store_tiles();
    __syncthreads();
    if (k0 + 32 < K) load_tiles(k0 + 32);  // overlap next-tile fetch with WMMA

    const int kbase0 = 8 * hi;
    bf16x16 afr[2];
#pragma unroll
    for (int mt = 0; mt < 2; ++mt) {
      const bf16_t* ap = &As[(wm * 32 + mt * 16 + lo) * 32 + kbase0];
      afr[mt] = load_frag(ap, ap + 16);
    }
    bf16x16 bfr[4];
#pragma unroll
    for (int nt = 0; nt < 4; ++nt) {
      const bf16_t* bp = &Bs[(wn * 64 + nt * 16 + lo) * 32 + kbase0];
      bfr[nt] = load_frag(bp, bp + 16);
    }
#pragma unroll
    for (int mt = 0; mt < 2; ++mt)
#pragma unroll
      for (int nt = 0; nt < 4; ++nt)
        acc[mt][nt] = __builtin_amdgcn_wmma_f32_16x16x32_bf16(
            false, afr[mt], false, bfr[nt], (short)0, acc[mt][nt], false, false);
    __syncthreads();
  }

  // D layout: VGPR i, lanes 0-15 -> row i, lanes 16-31 -> row i+8; col = lane&15
#pragma unroll
  for (int mt = 0; mt < 2; ++mt)
#pragma unroll
    for (int nt = 0; nt < 4; ++nt)
#pragma unroll
      for (int i = 0; i < 8; ++i) {
        int gr = blockM + wm * 32 + mt * 16 + i + 8 * hi;
        int gc = blockN + wn * 64 + nt * 16 + lo;
        if (gr < M && gc < Nn) {
          float vv = acc[mt][nt][i];
          if (bias) vv += bias[gc];
          C[cOff + (long long)gr * ldc + gc] = vv;
        }
      }
}

// ---------------------------------------------------------------------------
// Flash attention: per (n,h), 128 q-rows per block, 128-key tiles, d = 64.
// Scores never hit HBM (avoids 402 MB of traffic). bf16 WMMA, fp32 softmax.
// ---------------------------------------------------------------------------
__global__ __launch_bounds__(256) void fa_kernel(
    const float* __restrict__ Qm, const float* __restrict__ Km,
    const float* __restrict__ V, float* __restrict__ Out)
{
  constexpr int S = 1024, E = 768, H = 12, HD = 64;
  __shared__ __align__(16) bf16_t Qs[128 * 64];       // [qrow][d]
  __shared__ __align__(16) bf16_t Ks[128 * 64];       // [key][d]
  __shared__ __align__(16) bf16_t Vt[64 * 128];       // [d][key] (transposed)
  __shared__ __align__(16) bf16_t Ps[8 * 16 * 128];   // per-wave P pane [row][key]

  const int b = blockIdx.y;
  const int n = b / H, h = b % H;
  const int q0 = blockIdx.x * 128;
  const int tid = threadIdx.x;
  const int lane = tid & 31;
  const int w = tid >> 5;            // wave owns q rows [w*16, w*16+16)
  const int lo = lane & 15, hi = lane >> 4;
  const int kb0 = 8 * hi;

  // stage Q tile once: 2048 float4, 8 per thread, fully contiguous
  const float* qbase = Qm + (size_t)b * S * HD + (size_t)q0 * HD;
#pragma unroll
  for (int e = 0; e < 8; ++e) {
    int idx = tid + e * 256;
    float4 v = *(const float4*)(qbase + idx * 4);
    store_bf4(&Qs[idx * 4], v);
  }

  float m_i[8], l_i[8];
#pragma unroll
  for (int i = 0; i < 8; ++i) { m_i[i] = -1e30f; l_i[i] = 0.0f; }
  v8f o[4];
#pragma unroll
  for (int t = 0; t < 4; ++t) o[t] = zero_v8f();

  const float* kBatch = Km + (size_t)b * S * HD;
  const float* vBatch = V + (size_t)n * S * E + h * HD;

  for (int j0 = 0; j0 < S; j0 += 128) {
    // stage K tile [key][d] (contiguous) and V tile transposed [d][key]
    const float* kbase = kBatch + (size_t)j0 * HD;
#pragma unroll
    for (int e = 0; e < 8; ++e) {
      int idx = tid + e * 256;
      float4 v = *(const float4*)(kbase + idx * 4);
      store_bf4(&Ks[idx * 4], v);
    }
    const float* vbase = vBatch + (size_t)j0 * E;
#pragma unroll
    for (int e = 0; e < 8; ++e) {
      int idx = tid + e * 256;
      int t = idx >> 4, d = (idx & 15) << 2;
      float4 v = *(const float4*)(vbase + (size_t)t * E + d);
      Vt[(d + 0) * 128 + t] = f2bf(v.x);
      Vt[(d + 1) * 128 + t] = f2bf(v.y);
      Vt[(d + 2) * 128 + t] = f2bf(v.z);
      Vt[(d + 3) * 128 + t] = f2bf(v.w);
    }
    __syncthreads();

    // hint-prefetch next j-tile while this one computes (global_prefetch_b8)
    if (j0 + 128 < S) {
      __builtin_prefetch(kBatch + (size_t)(j0 + 128) * HD + tid * 32, 0, 1);
      __builtin_prefetch(vBatch + (size_t)(j0 + 128 + (tid >> 1)) * E + (tid & 1) * 32, 0, 1);
    }

    // S = (Q' K'^T) * 1/sqrt(hd) : hoist the 2 Q fragments over 8 n-tiles
    bf16x16 qa[2];
#pragma unroll
    for (int kkI = 0; kkI < 2; ++kkI) {
      const bf16_t* ap = &Qs[(w * 16 + lo) * 64 + kkI * 32 + kb0];
      qa[kkI] = load_frag(ap, ap + 16);
    }
    v8f s[8];
#pragma unroll
    for (int nt = 0; nt < 8; ++nt) {
      v8f acc = zero_v8f();
#pragma unroll
      for (int kkI = 0; kkI < 2; ++kkI) {
        const bf16_t* bp = &Ks[(nt * 16 + lo) * 64 + kkI * 32 + kb0];
        bf16x16 bfr = load_frag(bp, bp + 16);
        acc = __builtin_amdgcn_wmma_f32_16x16x32_bf16(
            false, qa[kkI], false, bfr, (short)0, acc, false, false);
      }
#pragma unroll
      for (int i = 0; i < 8; ++i) acc[i] *= 0.125f;  // 1/sqrt(64)
      s[nt] = acc;
    }

    // online softmax: row r = i + 8*hi lives in a 16-lane half -> shfl_xor 1..8
    float mnew[8], corr[8], rsum[8];
#pragma unroll
    for (int i = 0; i < 8; ++i) {
      float mx = s[0][i];
#pragma unroll
      for (int nt = 1; nt < 8; ++nt) mx = fmaxf(mx, s[nt][i]);
#pragma unroll
      for (int d = 1; d < 16; d <<= 1) mx = fmaxf(mx, __shfl_xor(mx, d, 32));
      mnew[i] = fmaxf(m_i[i], mx);
      corr[i] = __expf(m_i[i] - mnew[i]);
      m_i[i] = mnew[i];
      rsum[i] = 0.0f;
    }
#pragma unroll
    for (int nt = 0; nt < 8; ++nt)
#pragma unroll
      for (int i = 0; i < 8; ++i) {
        float p = __expf(s[nt][i] - mnew[i]);
        s[nt][i] = p;
        rsum[i] += p;
      }
#pragma unroll
    for (int i = 0; i < 8; ++i) {
#pragma unroll
      for (int d = 1; d < 16; d <<= 1) rsum[i] += __shfl_xor(rsum[i], d, 32);
      l_i[i] = l_i[i] * corr[i] + rsum[i];
    }
#pragma unroll
    for (int t = 0; t < 4; ++t)
#pragma unroll
      for (int i = 0; i < 8; ++i) o[t][i] *= corr[i];

    // reshape P: D-layout regs -> row-major bf16 pane (per-wave, in-order LDS)
#pragma unroll
    for (int nt = 0; nt < 8; ++nt)
#pragma unroll
      for (int i = 0; i < 8; ++i)
        Ps[w * 2048 + (i + 8 * hi) * 128 + nt * 16 + lo] = f2bf(s[nt][i]);
    __syncthreads();

    // O += P @ V : hoist the 4 P fragments over the 4 output tiles
    bf16x16 pa[4];
#pragma unroll
    for (int kc = 0; kc < 4; ++kc) {
      const bf16_t* ap = &Ps[w * 2048 + lo * 128 + kc * 32 + kb0];
      pa[kc] = load_frag(ap, ap + 16);
    }
#pragma unroll
    for (int ot = 0; ot < 4; ++ot) {
#pragma unroll
      for (int kc = 0; kc < 4; ++kc) {
        const bf16_t* bp = &Vt[(ot * 16 + lo) * 128 + kc * 32 + kb0];
        bf16x16 bfr = load_frag(bp, bp + 16);
        o[ot] = __builtin_amdgcn_wmma_f32_16x16x32_bf16(
            false, pa[kc], false, bfr, (short)0, o[ot], false, false);
      }
    }
    __syncthreads();
  }

  // epilogue: O / l, scatter back to [N,S,E] head slot
  float* obase = Out + (size_t)n * S * E + (size_t)(q0 + w * 16) * E + h * HD;
#pragma unroll
  for (int ot = 0; ot < 4; ++ot)
#pragma unroll
    for (int i = 0; i < 8; ++i) {
      int row = i + 8 * hi;
      int col = ot * 16 + lo;
      obase[(size_t)row * E + col] = o[ot][i] / l_i[i];
    }
}

// ---------------------------------------------------------------------------
// Host launcher
// ---------------------------------------------------------------------------
static void launch_gemm(const float* A, const float* B, const float* bias, float* C,
                        int M, int Nn, int K, int lda, int ldb, int ldc, int transB,
                        int innerCount,
                        long long aO, long long aI, long long bO, long long bI,
                        long long cO, long long cI, int batch, hipStream_t s)
{
  dim3 grid((Nn + 127) / 128, (M + 127) / 128, batch);
  gemm_bf16_kernel<<<grid, dim3(256), 0, s>>>(A, B, bias, C, M, Nn, K, lda, ldb, ldc,
                                              transB, innerCount, aO, aI, bO, bI, cO, cI);
}

extern "C" void kernel_launch(void* const* d_in, const int* in_sizes, int n_in,
                              void* d_out, int out_size, void* d_ws, size_t ws_size,
                              hipStream_t stream) {
  constexpr int Nb = 8, S = 1024, E = 768, R = 384, H = 12, HD = 64;
  constexpr int MM = Nb * S;  // 8192 token rows

  const float* HS  = (const float*)d_in[0];
  const float* U_Q = (const float*)d_in[1];
  const float* V_Q = (const float*)d_in[2];
  const float* U_K = (const float*)d_in[3];
  const float* V_K = (const float*)d_in[4];
  const float* U_V = (const float*)d_in[5];
  const float* V_V = (const float*)d_in[6];
  const float* bq  = (const float*)d_in[7];
  const float* bk  = (const float*)d_in[8];
  const float* bv  = (const float*)d_in[9];
  const float* M_U = (const float*)d_in[10];
  const float* M_V = (const float*)d_in[11];
  float* out = (float*)d_out;

  float* ws  = (float*)d_ws;
  float* tmp = ws;                          // MM*R
  float* qb  = tmp + (size_t)MM * R;        // MM*E
  float* kb  = qb  + (size_t)MM * E;        // MM*E
  float* vb  = kb  + (size_t)MM * E;        // MM*E
  float* Qm  = vb  + (size_t)MM * E;        // Nb*H*S*HD = 96*1024*64
  float* Km  = qb;                          // reuse qb once Qm is done

  const long long sNE  = (long long)S * E;       // per-n stride in [N,S,E]
  const long long bOut = (long long)H * S * HD;  // per-n stride in [N*H,S,HD]
  const long long bIn  = (long long)S * HD;      // per-head stride in [N*H,S,HD]

  // ---- Q chain: q = HS @ V_Q @ U_Q + bq ; Q' = q_head @ M_U
  launch_gemm(HS,  V_Q, nullptr, tmp, MM, R, E, E, R, R, 0, 1, 0,0, 0,0, 0,0, 1, stream);
  launch_gemm(tmp, U_Q, bq,      qb,  MM, E, R, R, E, E, 0, 1, 0,0, 0,0, 0,0, 1, stream);
  launch_gemm(qb,  M_U, nullptr, Qm,  S, HD, HD, E, HD, HD, 0,
              H, sNE, HD, 0, 0, bOut, bIn, Nb * H, stream);

  // ---- K chain: k = HS @ V_K @ U_K + bk ; K' = k_head @ M_V^T  (into qb)
  launch_gemm(HS,  V_K, nullptr, tmp, MM, R, E, E, R, R, 0, 1, 0,0, 0,0, 0,0, 1, stream);
  launch_gemm(tmp, U_K, bk,      kb,  MM, E, R, R, E, E, 0, 1, 0,0, 0,0, 0,0, 1, stream);
  launch_gemm(kb,  M_V, nullptr, Km,  S, HD, HD, E, HD, HD, 1,
              H, sNE, HD, 0, 0, bOut, bIn, Nb * H, stream);

  // ---- V chain: v = HS @ V_V @ U_V + bv
  launch_gemm(HS,  V_V, nullptr, tmp, MM, R, E, E, R, R, 0, 1, 0,0, 0,0, 0,0, 1, stream);
  launch_gemm(tmp, U_V, bv,      vb,  MM, E, R, R, E, E, 0, 1, 0,0, 0,0, 0,0, 1, stream);

  // ---- Flash attention over (n,h) batches
  dim3 grid(S / 128, Nb * H);
  fa_kernel<<<grid, dim3(256), 0, stream>>>(Qm, Km, vb, out);
}
